// VectorQuantize_9861244912400
// MI455X (gfx1250) — matrix operand order repaired and checked by reading, not compiled, not run on hardware
//
#include <hip/hip_runtime.h>
#include <hip/hip_bf16.h>

// Problem sizes (fixed by the reference)
#define B_  8
#define D_  512
#define T_  2048
#define K_  8192
#define BT_ (B_ * T_)
#define QN_ (B_ * D_ * T_)
#define NKT (K_ / 16)

// Tiling
#define SA_T 128          // tokens per block (8 waves x 16-row WMMA tile)
#define SROW 520          // padded LDS row stride in halfs (= 512 + 8; TDM pad reproduces this)
#define TILE_BYTES ((size_t)16 * D_ * 2)   // one 16-code bf16 tile in global

#ifndef __has_builtin
#define __has_builtin(x) 0
#endif
#if __has_builtin(__builtin_amdgcn_tensor_load_to_lds)
#define HAVE_TDM 1
#else
#define HAVE_TDM 0
#endif

typedef __attribute__((ext_vector_type(4)))  __bf16   bf16x4;
typedef __attribute__((ext_vector_type(8)))  __bf16   bf16x8;
typedef __attribute__((ext_vector_type(16))) __bf16   bf16x16;
typedef __attribute__((ext_vector_type(8)))  float    f32x8;
typedef __attribute__((ext_vector_type(4)))  unsigned u32x4;
typedef __attribute__((ext_vector_type(8)))  int      i32x8;
typedef __attribute__((ext_vector_type(4)))  int      i32x4;

static __device__ __forceinline__ bf16x16 cat16(bf16x8 lo, bf16x8 hi) {
  return __builtin_shufflevector(lo, hi, 0,1,2,3,4,5,6,7,8,9,10,11,12,13,14,15);
}

#if HAVE_TDM
// One TDM descriptor: load a 16-row x 512-half bf16 tile into LDS with a
// 16B pad after every 1024B row (-> LDS row stride = SROW halfs).
static __device__ __forceinline__ void tdm_load_tile(unsigned long long gaddr,
                                                     unsigned lds_off) {
  u32x4 g0;
  g0[0] = 1u;                                          // count=1, user descriptor
  g0[1] = lds_off;                                     // LDS byte address
  g0[2] = (unsigned)(gaddr & 0xffffffffull);           // global_addr[31:0]
  g0[3] = (unsigned)((gaddr >> 32) & 0x1ffffffull)     // global_addr[56:32]
        | (2u << 30);                                  // type = 2 ("image")
  i32x8 g1;
  g1[0] = (int)((1u << 16)       // data_size = 2 bytes
              | (1u << 20)       // pad_enable
              | (7u << 22)       // pad_interval: every 256 DWORDs (1024B row)
              | (3u << 25));     // pad_amount: 4 DWORDs (16B) -> SROW stride
  g1[1] = (int)(512u << 16);     // tensor_dim0 = 512 elements
  g1[2] = (int)(16u  << 16);     // tensor_dim1 = 16 rows (tile-local tensor)
  g1[3] = (int)(512u << 16);     // tile_dim0 = 512
  g1[4] = 16;                    // tile_dim1 = 16, tile_dim2 = 0
  g1[5] = 512;                   // tensor_dim0_stride = 512 elements
  g1[6] = 0;
  g1[7] = 0;
  i32x4 gz = {0, 0, 0, 0};       // 2-D tensor: groups 2/3 unused
#if __clang_major__ >= 23
  i32x8 gz8 = {0, 0, 0, 0, 0, 0, 0, 0};
  __builtin_amdgcn_tensor_load_to_lds(g0, g1, gz, gz, gz8, 0);
#else
  __builtin_amdgcn_tensor_load_to_lds(g0, g1, gz, gz, 0);
#endif
}
#endif

// ---------------------------------------------------------------------------
// (1) zero the commit-loss accumulator in workspace (deterministic each call)
__global__ void vq_init(float* __restrict__ loss_acc) { *loss_acc = 0.0f; }

// ---------------------------------------------------------------------------
// (2) codebook fp32 -> bf16 (row-major, WMMA-friendly) + e2[k] = ||e_k||^2
__global__ __launch_bounds__(128)
void vq_prep(const float* __restrict__ cb, __bf16* __restrict__ cbb,
             float* __restrict__ e2) {
  const int row = blockIdx.x;
  const int tid = threadIdx.x;                       // 128 threads, 4 floats each
  const float4 v = ((const float4*)(cb + (size_t)row * D_))[tid];
  bf16x4 o;
  o[0] = (__bf16)v.x; o[1] = (__bf16)v.y; o[2] = (__bf16)v.z; o[3] = (__bf16)v.w;
  ((bf16x4*)(cbb + (size_t)row * D_))[tid] = o;
  float s = v.x*v.x + v.y*v.y + v.z*v.z + v.w*v.w;
  #pragma unroll
  for (int off = 16; off; off >>= 1) s += __shfl_down(s, off, 32);
  __shared__ float ps[4];
  if ((tid & 31) == 0) ps[tid >> 5] = s;
  __syncthreads();
  if (tid == 0) e2[row] = ps[0] + ps[1] + ps[2] + ps[3];
}

// ---------------------------------------------------------------------------
// (3) main kernel: per block, 128 tokens vs all 8192 codes via bf16 WMMA.
//     argmin_k( e2[k] - 2 * x.e_k );  B tiles DMA'd by the Tensor Data Mover,
//     double-buffered, one barrier per k-tile.
__global__ __launch_bounds__(256)
void vq_main(const float* __restrict__ x, const __bf16* __restrict__ cbb,
             const float* __restrict__ e2g, int* __restrict__ idx_out) {
  extern __shared__ __bf16 smem[];
  __bf16* sA  = smem;                                  // [SA_T][SROW] tokens (bf16)
  __bf16* sB0 = smem + (size_t)SA_T * SROW;            // [16][SROW]  tile buf 0
  __bf16* sB1 = sB0 + (size_t)16 * SROW;               // [16][SROW]  tile buf 1

  const int tid  = threadIdx.x;
  const int wave = tid >> 5;
  const int lane = tid & 31;
  const int bt0  = blockIdx.x * SA_T;                  // 128 tokens, same batch b
  const int b    = bt0 / T_;
  const int t0   = bt0 % T_;

  // ---- stage A: transpose x[b, d, t0..t0+127] -> sA[tt][d] as bf16 ----
  {
    const int tt = tid & 127;                          // contiguous along T: coalesced
    const int dg = tid >> 7;                           // 0..1
    const float* xb = x + (size_t)b * D_ * T_ + t0;
    for (int d = dg; d < D_; d += 2)
      sA[tt * SROW + d] = (__bf16)xb[(size_t)d * T_ + tt];
  }

  const unsigned long long gbase = (unsigned long long)(uintptr_t)cbb;
  const unsigned ldsB[2] = { (unsigned)(uintptr_t)sB0, (unsigned)(uintptr_t)sB1 };
  (void)gbase; (void)ldsB;

  // WMMA fragment geometry (ISA 7.12.2)
  const int m    = lane & 15;                          // A row / B column / C column
  const int lo   = (lane & 16) ? 8 : 0;                // A: K chunk select
  const int bsel = (lane & 16) ? 16 : 0;               // unused; B uses lo, same k order as A
  const __bf16* aRow = sA + (size_t)(wave * 16 + m) * SROW;

  float minv[8];
  int   mini[8];
  #pragma unroll
  for (int v = 0; v < 8; ++v) { minv[v] = 3.4e38f; mini[v] = 0; }

  // ---- prologue: stage tile 0 into buffer 0 ----
#if HAVE_TDM
  if (wave == 0) tdm_load_tile(gbase, ldsB[0]);
#else
  {
    const int row = tid >> 4, seg = tid & 15;
    const bf16x8* g = (const bf16x8*)(cbb + (size_t)row * D_ + seg * 32);
    bf16x8* d = (bf16x8*)(sB0 + (size_t)row * SROW + seg * 32);
    d[0] = g[0]; d[1] = g[1]; d[2] = g[2]; d[3] = g[3];
  }
#endif

  for (int kt = 0; kt < NKT; ++kt) {
#if HAVE_TDM
    if (wave == 0) __builtin_amdgcn_s_wait_tensorcnt(0);  // buf[kt&1] ready
#endif
    __syncthreads();   // staging visible to all; compute of kt-1 finished

    // ---- kick DMA for next tile into the other buffer (overlaps compute) ----
    if (kt + 1 < NKT) {
#if HAVE_TDM
      if (wave == 0)
        tdm_load_tile(gbase + (unsigned long long)(kt + 1) * TILE_BYTES,
                      ldsB[(kt + 1) & 1]);
#else
      const int row = tid >> 4, seg = tid & 15;
      const bf16x8* g = (const bf16x8*)(cbb + (size_t)(kt + 1) * 16 * D_
                                            + (size_t)row * D_ + seg * 32);
      bf16x8* d = (bf16x8*)(((kt + 1) & 1 ? sB1 : sB0)
                            + (size_t)row * SROW + seg * 32);
      d[0] = g[0]; d[1] = g[1]; d[2] = g[2]; d[3] = g[3];
#endif
    }

    const float e2v = e2g[kt * 16 + m];                // L2-resident, issued early
    const __bf16* bRow = ((kt & 1) ? sB1 : sB0) + (size_t)m * SROW;

    f32x8 c = {};                                      // x . e accumulator (f32)
    #pragma unroll
    for (int kk = 0; kk < 16; ++kk) {                  // D = 512 = 16 x 32
      const int kb = kk * 32;
      bf16x8 a0 = *(const bf16x8*)(aRow + kb + lo);
      bf16x8 a1 = *(const bf16x8*)(aRow + kb + 16 + lo);
      bf16x8 b0 = *(const bf16x8*)(bRow + kb + lo);
      bf16x8 b1 = *(const bf16x8*)(bRow + kb + 16 + lo);
      c = __builtin_amdgcn_wmma_f32_16x16x32_bf16(
              false, cat16(a0, a1), false, cat16(b0, b1),
              (short)0, c, false, false);
    }
    const int code = kt * 16 + m;
    #pragma unroll
    for (int v = 0; v < 8; ++v) {
      const float s = fmaf(-2.0f, c[v], e2v);
      if (s < minv[v]) { minv[v] = s; mini[v] = code; }
    }
  }

  // ---- argmin across 16 column-lanes (xor masks stay inside each half) ----
  #pragma unroll
  for (int v = 0; v < 8; ++v) {
    float mv = minv[v]; int mi = mini[v];
    #pragma unroll
    for (int msk = 1; msk < 16; msk <<= 1) {
      const float ov = __shfl_xor(mv, msk, 32);
      const int   oi = __shfl_xor(mi, msk, 32);
      if (ov < mv || (ov == mv && oi < mi)) { mv = ov; mi = oi; }
    }
    if ((lane & 15) == 0) {
      const int tt = wave * 16 + v + ((lane & 16) ? 8 : 0);  // C row layout
      idx_out[bt0 + tt] = mi;
    }
  }
}

// ---------------------------------------------------------------------------
// (4) gather quantized = codebook[idx] into [B,D,T] (coalesced along T),
//     accumulate commit loss sum of (q - x)^2
__global__ __launch_bounds__(256)
void vq_gather(const float* __restrict__ x, const float* __restrict__ cb,
               const int* __restrict__ idx, float* __restrict__ q,
               float* __restrict__ loss_acc) {
  const int t = blockIdx.x * 256 + threadIdx.x;
  const int d = blockIdx.y;
  const int b = blockIdx.z;
  const int code = idx[b * T_ + t];
  const float qv = cb[(size_t)code * D_ + d];          // codebook resident in L2
  const size_t o = ((size_t)b * D_ + d) * T_ + t;
  const float diff = qv - x[o];
  q[o] = qv;
  float s = diff * diff;
  #pragma unroll
  for (int off = 16; off; off >>= 1) s += __shfl_down(s, off, 32);
  __shared__ float ws[8];
  if ((threadIdx.x & 31) == 0) ws[threadIdx.x >> 5] = s;
  __syncthreads();
  if (threadIdx.x == 0) {
    float tot = 0.0f;
    #pragma unroll
    for (int w = 0; w < 8; ++w) tot += ws[w];
    atomicAdd(loss_acc, tot);                          // global_atomic_add_f32
  }
}

// ---------------------------------------------------------------------------
// (5) finalize: commit_loss = 0.25 * mean
__global__ void vq_finalize(const float* __restrict__ acc, float* __restrict__ out) {
  *out = 0.25f * (*acc) / (float)((long long)B_ * T_ * D_);
}

// ---------------------------------------------------------------------------
extern "C" void kernel_launch(void* const* d_in, const int* in_sizes, int n_in,
                              void* d_out, int out_size, void* d_ws, size_t ws_size,
                              hipStream_t stream) {
  (void)in_sizes; (void)n_in; (void)out_size; (void)ws_size;
  const float* x  = (const float*)d_in[0];             // [B, D, T] fp32
  const float* cb = (const float*)d_in[1];             // [K, D]    fp32

  float* qout  = (float*)d_out;                        // [B, D, T]
  int*   idxo  = (int*)(qout + QN_);                   // [B, T] int32
  float* losso = qout + QN_ + BT_;                     // scalar

  char*   ws   = (char*)d_ws;
  __bf16* cbb  = (__bf16*)ws;                                        // 8 MB
  float*  e2   = (float*)(ws + (size_t)K_ * D_ * sizeof(__bf16));    // 32 KB
  float*  lacc = (float*)(ws + (size_t)K_ * D_ * sizeof(__bf16) + (size_t)K_ * sizeof(float));

  const size_t smem = (size_t)(SA_T + 32) * SROW * sizeof(__bf16);   // A + 2x B buffers
  (void)hipFuncSetAttribute((const void*)vq_main,
                            hipFuncAttributeMaxDynamicSharedMemorySize, (int)smem);

  vq_init<<<1, 1, 0, stream>>>(lacc);
  vq_prep<<<K_, 128, 0, stream>>>(cb, cbb, e2);
  vq_main<<<BT_ / SA_T, 256, smem, stream>>>(x, cbb, e2, idxo);
  vq_gather<<<dim3(T_ / 256, D_, B_), 256, 0, stream>>>(x, cb, idxo, qout, lacc);
  vq_finalize<<<1, 1, 0, stream>>>(lacc, losso);
}
